// UR4RecV2MoE_14594298872603
// MI455X (gfx1250) — compile-verified, matrix operand order checked by reading, not run on hardware
//
#include <hip/hip_runtime.h>

// ---------------------------------------------------------------------------
// Problem dims (compile-time constants from the reference)
// ---------------------------------------------------------------------------
#define B_  4096
#define D_  256
#define VD_ 512
#define TD_ 384
#define L_  50
#define AH_ 128
#define BL_ (B_ * L_)   // 204800

typedef __attribute__((ext_vector_type(16))) __bf16 v16bf;
typedef __attribute__((ext_vector_type(8)))  float  v8f;

union Frag8 { unsigned int u[8]; v16bf v; };

// ---------------------------------------------------------------------------
// f32 -> bf16 conversion (weights staged once per launch)
// ---------------------------------------------------------------------------
__global__ void f32_to_bf16_kernel(const float* __restrict__ src,
                                   __bf16* __restrict__ dst, int n) {
  int i = blockIdx.x * blockDim.x + threadIdx.x;
  if (i < n) dst[i] = (__bf16)src[i];
}

// ---------------------------------------------------------------------------
// Generic GEMM: C[M,N] = A[M,K] @ W[N,K]^T + bias[N]  (optional ReLU)
// A: f32 -> bf16 while staging (VALU path, register-pipelined).
// W: bf16 -> LDS via CDNA5 async copy (global_load_async_to_lds_b64,
//    ASYNCcnt-tracked, overlaps the WMMA block).
// Double-buffered LDS, block tile 128x128x32, 256 threads (8 waves, 2Mx4N),
// each wave: 4x2 tiles of 16x16 -> 8x v_wmma_f32_16x16x32_bf16 per K-step.
// Requires: M % 128 == 0, K % 32 == 0. N guarded (zero effect rows clamped).
// ---------------------------------------------------------------------------
#define LDSK 36   // padded bf16 row stride: 18 dwords, gcd(18,64)=2 -> no LDS bank conflicts

__global__ __launch_bounds__(256) void gemm_bf16_kernel(
    const float* __restrict__ A, const __bf16* __restrict__ W,
    const float* __restrict__ bias, float* __restrict__ C,
    int M, int N, int K, int relu)
{
  __shared__ alignas(16) __bf16 As[2][128 * LDSK];
  __shared__ alignas(16) __bf16 Ws[2][128 * LDSK];

  const int tid  = threadIdx.x;
  const int lane = tid & 31;
  const int wave = tid >> 5;
  const int wm   = wave >> 2;   // 0..1 (64 rows each)
  const int wn   = wave & 3;    // 0..3 (32 cols each)
  const int r16  = lane & 15;
  const int hi   = lane >> 4;
  const int m0 = blockIdx.x * 128;
  const int n0 = blockIdx.y * 128;

  // staging coordinates: 128x32 tile, 256 threads, 4 units each
  const int srow = tid >> 3;          // base row (i-th unit adds 32)
  const int scol = (tid & 7) << 2;    // 4-element column group

  v8f acc[4][2];
  #pragma unroll
  for (int i = 0; i < 4; ++i)
    #pragma unroll
    for (int j = 0; j < 2; ++j) acc[i][j] = (v8f)(0.0f);

  float4 rA[4];

  // issue A-tile global loads into registers (no wait here)
  auto issueA = [&](int kt) {
    #pragma unroll
    for (int i = 0; i < 4; ++i)
      rA[i] = *(const float4*)(A + (size_t)(m0 + srow + i * 32) * K + kt + scol);
  };
  // convert + store A registers into LDS buffer
  auto storeA = [&](int buf) {
    #pragma unroll
    for (int i = 0; i < 4; ++i) {
      union { __bf16 h[4]; uint2 u; } pk;
      pk.h[0] = (__bf16)rA[i].x; pk.h[1] = (__bf16)rA[i].y;
      pk.h[2] = (__bf16)rA[i].z; pk.h[3] = (__bf16)rA[i].w;
      *(uint2*)&As[buf][(srow + i * 32) * LDSK + scol] = pk.u;
    }
  };
  // async-copy W tile (bf16, 8 B per lane) straight into LDS; ASYNCcnt-tracked
  auto issueW = [&](int kt, int buf) {
    #pragma unroll
    for (int i = 0; i < 4; ++i) {
      int row = srow + i * 32;
      int gr  = (n0 + row < N) ? (n0 + row) : (N - 1);   // clamp: no OOB reads
      const void* g = (const void*)(W + (size_t)gr * K + kt + scol);
      unsigned ldsoff = (unsigned)(uintptr_t)&Ws[buf][row * LDSK + scol];
      asm volatile("global_load_async_to_lds_b64 %0, %1, off"
                   :: "v"(ldsoff), "v"(g) : "memory");
    }
  };

  const int nk = K >> 5;

  // prologue: stage tile 0
  issueA(0);
  issueW(0, 0);
  storeA(0);                                     // waits its own loadcnt
  asm volatile("s_wait_asynccnt 0" ::: "memory");
  __syncthreads();

  for (int t = 0; t < nk; ++t) {
    const int cur = t & 1, nxt = cur ^ 1;
    const bool more = (t + 1) < nk;
    if (more) {                                  // overlap next tile with WMMA
      issueA((t + 1) << 5);
      issueW((t + 1) << 5, nxt);
    }

    // --- fragments (ISA 7.12.2 16-bit A layout: K pairs per VGPR) ---
    Frag8 af[4], bfr[2];
    #pragma unroll
    for (int mt = 0; mt < 4; ++mt) {
      int row = wm * 64 + mt * 16 + r16;
      #pragma unroll
      for (int v = 0; v < 8; ++v) {
        int k = ((v >> 2) << 4) + (hi << 3) + ((v & 3) << 1);
        af[mt].u[v] = *(const unsigned int*)&As[cur][row * LDSK + k];
      }
    }
    #pragma unroll
    for (int nt = 0; nt < 2; ++nt) {
      int row = wn * 32 + nt * 16 + r16;
      #pragma unroll
      for (int v = 0; v < 8; ++v) {
        int k = ((v >> 2) << 4) + (hi << 3) + ((v & 3) << 1);
        bfr[nt].u[v] = *(const unsigned int*)&Ws[cur][row * LDSK + k];
      }
    }
    #pragma unroll
    for (int mt = 0; mt < 4; ++mt)
      #pragma unroll
      for (int nt = 0; nt < 2; ++nt)
        acc[mt][nt] = __builtin_amdgcn_wmma_f32_16x16x32_bf16(
            false, af[mt].v, false, bfr[nt].v, (short)0, acc[mt][nt], false, false);

    if (more) {
      storeA(nxt);                               // consume pipelined registers
      asm volatile("s_wait_asynccnt 0" ::: "memory");
    }
    __syncthreads();
  }

  // --- epilogue: C layout (lanes 0-15: N=lane, M=r; lanes 16-31: N=lane-16, M=8+r)
  #pragma unroll
  for (int mt = 0; mt < 4; ++mt) {
    #pragma unroll
    for (int nt = 0; nt < 2; ++nt) {
      int gn = n0 + wn * 32 + nt * 16 + r16;
      if (gn < N) {
        float bv  = bias[gn];
        int   gmb = m0 + wm * 64 + mt * 16 + hi * 8;
        const float* ap = (const float*)&acc[mt][nt];
        #pragma unroll
        for (int r = 0; r < 8; ++r) {
          float val = ap[r] + bv;
          if (relu) val = fmaxf(val, 0.0f);
          C[(size_t)(gmb + r) * N + gn] = val;
        }
      }
    }
  }
}

// ---------------------------------------------------------------------------
// LayerNorm (one wave per row), optional ReLU. D % 32 == 0, D <= 768.
// ---------------------------------------------------------------------------
__global__ __launch_bounds__(256) void ln_kernel(
    const float* __restrict__ x, const float* __restrict__ g,
    const float* __restrict__ bta, float* __restrict__ y,
    int M, int D, int relu)
{
  int wid = blockIdx.x * 8 + (threadIdx.x >> 5);
  if (wid >= M) return;
  int lane = threadIdx.x & 31;
  int nper = D >> 5;
  const float* xr = x + (size_t)wid * D;
  float s = 0.f, s2 = 0.f;
  for (int i = 0; i < nper; ++i) {
    float t = xr[lane + (i << 5)];
    s += t; s2 += t * t;
  }
  #pragma unroll
  for (int off = 16; off; off >>= 1) {
    s  += __shfl_xor(s,  off, 32);
    s2 += __shfl_xor(s2, off, 32);
  }
  float mean = s / (float)D;
  float var  = s2 / (float)D - mean * mean;
  float rstd = rsqrtf(var + 1e-5f);
  float* yr = y + (size_t)wid * D;
  for (int i = 0; i < nper; ++i) {
    int c = lane + (i << 5);
    float t = (xr[c] - mean) * rstd * g[c] + bta[c];
    if (relu) t = fmaxf(t, 0.f);
    yr[c] = t;
  }
}

// ---------------------------------------------------------------------------
// VisualExpert fused attention: sc = kh.q/sqrt(AH); softmax over L; agg = aw@mem_vis
// one block (128 thr, 4 waves) per batch row
// ---------------------------------------------------------------------------
__global__ __launch_bounds__(128) void ve_attn_kernel(
    const float* __restrict__ qh, const float* __restrict__ kh,
    const float* __restrict__ mv, float* __restrict__ agg)
{
  int b = blockIdx.x;
  int tid = threadIdx.x, lane = tid & 31, wave = tid >> 5;
  __shared__ float sc[64];
  const float* q = qh + (size_t)b * AH_;
  for (int l = wave; l < L_; l += 4) {
    const float* kr = kh + ((size_t)b * L_ + l) * AH_;
    float p = 0.f;
    #pragma unroll
    for (int i = 0; i < 4; ++i) p += kr[lane + (i << 5)] * q[lane + (i << 5)];
    #pragma unroll
    for (int off = 16; off; off >>= 1) p += __shfl_xor(p, off, 32);
    if (lane == 0) sc[l] = p * 0.0883883476483184f;  // 1/sqrt(128)
  }
  __syncthreads();
  if (wave == 0) {
    float v0 = (lane < L_)      ? sc[lane]      : -3.4e38f;
    float v1 = (lane + 32 < L_) ? sc[lane + 32] : -3.4e38f;
    float m = fmaxf(v0, v1);
    #pragma unroll
    for (int off = 16; off; off >>= 1) m = fmaxf(m, __shfl_xor(m, off, 32));
    float e0 = (lane < L_)      ? __expf(v0 - m) : 0.f;
    float e1 = (lane + 32 < L_) ? __expf(v1 - m) : 0.f;
    float s = e0 + e1;
    #pragma unroll
    for (int off = 16; off; off >>= 1) s += __shfl_xor(s, off, 32);
    float inv = 1.f / s;
    if (lane < L_)      sc[lane]      = e0 * inv;
    if (lane + 32 < L_) sc[lane + 32] = e1 * inv;
  }
  __syncthreads();
  const float* mvb = mv + (size_t)b * L_ * D_;
  float a0 = 0.f, a1 = 0.f;
  for (int l = 0; l < L_; ++l) {
    float wv = sc[l];
    a0 += wv * mvb[l * D_ + tid];
    a1 += wv * mvb[l * D_ + tid + 128];
  }
  agg[(size_t)b * D_ + tid]       = a0;
  agg[(size_t)b * D_ + tid + 128] = a1;
}

// ---------------------------------------------------------------------------
// SemanticExpert MHA core: 4 heads, hd=64, Lq=1, Lk=50.
// one block (128 thr) per batch row; one wave per head.
// ---------------------------------------------------------------------------
__global__ __launch_bounds__(128) void se_attn_kernel(
    const float* __restrict__ qp, const float* __restrict__ kp,
    const float* __restrict__ vp, float* __restrict__ o)
{
  int b = blockIdx.x, tid = threadIdx.x, lane = tid & 31, h = tid >> 5;
  __shared__ float sc[4][52];
  int d = h * 64 + lane * 2;
  const float* qb = qp + (size_t)b * D_;
  float qx = qb[d], qy = qb[d + 1];
  const float* kb = kp + (size_t)b * L_ * D_;
  for (int l = 0; l < L_; ++l) {
    float p = qx * kb[l * D_ + d] + qy * kb[l * D_ + d + 1];
    #pragma unroll
    for (int off = 16; off; off >>= 1) p += __shfl_xor(p, off, 32);
    if (lane == 0) sc[h][l] = p * 0.125f;  // 1/sqrt(64)
  }
  __syncthreads();
  float v0 = (lane < L_)      ? sc[h][lane]      : -3.4e38f;
  float v1 = (lane + 32 < L_) ? sc[h][lane + 32] : -3.4e38f;
  float m = fmaxf(v0, v1);
  #pragma unroll
  for (int off = 16; off; off >>= 1) m = fmaxf(m, __shfl_xor(m, off, 32));
  float e0 = (lane < L_)      ? __expf(v0 - m) : 0.f;
  float e1 = (lane + 32 < L_) ? __expf(v1 - m) : 0.f;
  float s = e0 + e1;
  #pragma unroll
  for (int off = 16; off; off >>= 1) s += __shfl_xor(s, off, 32);
  float inv = 1.f / s;
  if (lane < L_)      sc[h][lane]      = e0 * inv;
  if (lane + 32 < L_) sc[h][lane + 32] = e1 * inv;
  __syncthreads();
  const float* vb = vp + (size_t)b * L_ * D_;
  float ax = 0.f, ay = 0.f;
  for (int l = 0; l < L_; ++l) {
    float wv = sc[h][l];
    ax += wv * vb[l * D_ + d];
    ay += wv * vb[l * D_ + d + 1];
  }
  o[(size_t)b * D_ + d]     = ax;
  o[(size_t)b * D_ + d + 1] = ay;
}

// ---------------------------------------------------------------------------
// logits = x@w[2,K]^T + b; out = softmax(logits)  (one wave per row)
// ---------------------------------------------------------------------------
__global__ __launch_bounds__(256) void dual_softmax_kernel(
    const float* __restrict__ x, const float* __restrict__ w,
    const float* __restrict__ bias, float* __restrict__ out, int M, int K)
{
  int wid = blockIdx.x * 8 + (threadIdx.x >> 5);
  if (wid >= M) return;
  int lane = threadIdx.x & 31;
  const float* xr = x + (size_t)wid * K;
  float d0 = 0.f, d1 = 0.f;
  for (int c = lane; c < K; c += 32) {
    float t = xr[c];
    d0 += t * w[c];
    d1 += t * w[K + c];
  }
  #pragma unroll
  for (int off = 16; off; off >>= 1) {
    d0 += __shfl_xor(d0, off, 32);
    d1 += __shfl_xor(d1, off, 32);
  }
  if (lane == 0) {
    float l0 = d0 + bias[0], l1 = d1 + bias[1];
    float m = fmaxf(l0, l1);
    float e0 = __expf(l0 - m), e1 = __expf(l1 - m);
    float inv = 1.f / (e0 + e1);
    out[wid * 2]     = e0 * inv;
    out[wid * 2 + 1] = e1 * inv;
  }
}

// ---------------------------------------------------------------------------
// concat helpers and final combine
// ---------------------------------------------------------------------------
__global__ void concat3_kernel(const float* __restrict__ a, const float* __restrict__ b,
                               const float* __restrict__ c, float* __restrict__ out, int M) {
  int idx = blockIdx.x * blockDim.x + threadIdx.x;
  if (idx >= M * 768) return;
  int row = idx / 768, col = idx % 768;
  float v = (col < 256) ? a[row * 256 + col]
          : (col < 512) ? b[row * 256 + col - 256]
                        : c[row * 256 + col - 512];
  out[idx] = v;
}

__global__ void concat2_kernel(const float* __restrict__ a, const float* __restrict__ b,
                               float* __restrict__ out, int M) {
  int idx = blockIdx.x * blockDim.x + threadIdx.x;
  if (idx >= M * 512) return;
  int row = idx / 512, col = idx % 512;
  out[idx] = (col < 256) ? a[row * 256 + col] : b[row * 256 + col - 256];
}

__global__ void combine_kernel(const float* __restrict__ seq, const float* __restrict__ fused,
                               const float* __restrict__ visn, const float* __restrict__ semn,
                               const float* __restrict__ rw, const float* __restrict__ ag,
                               const float* __restrict__ gw, float* __restrict__ out, int total) {
  int idx = blockIdx.x * blockDim.x + threadIdx.x;
  if (idx >= total) return;
  int b = idx >> 8;  // /256
  float moe  = rw[2 * b] * visn[idx] + rw[2 * b + 1] * semn[idx];
  float comb = ag[2 * b] * fused[idx] + ag[2 * b + 1] * moe;
  out[idx] = seq[idx] + gw[0] * comb;
}

// ---------------------------------------------------------------------------
// Host orchestration
// ---------------------------------------------------------------------------
// input index map (setup_inputs dict order: 5 tensors, then params in order)
enum {
  I_SEQ = 0, I_TGT_EMB, I_TGT_VIS, I_MEM_VIS, I_MEM_TXT,
  P_VP_W, P_VP_B, P_TP_W, P_TP_B,
  P_VE_AQ_W, P_VE_AQ_B, P_VE_AK_W, P_VE_AK_B, P_VE_OUT_W, P_VE_OUT_B, P_VE_LN_G, P_VE_LN_B,
  P_SE_Q_W, P_SE_Q_B, P_SE_K_W, P_SE_K_B, P_SE_V_W, P_SE_V_B,
  P_SE_MHA_IN_W, P_SE_MHA_IN_B, P_SE_MHA_OUT_W, P_SE_MHA_OUT_B,
  P_SE_LN_G, P_SE_LN_B, P_SE_OUT_W, P_SE_OUT_B,
  P_CF_VQ_W, P_CF_VQ_B, P_CF_VK_W, P_CF_VK_B, P_CF_VV_W, P_CF_VV_B,
  P_CF_VMHA_IN_W, P_CF_VMHA_IN_B, P_CF_VMHA_OUT_W, P_CF_VMHA_OUT_B, P_CF_VO_W, P_CF_VO_B,
  P_CF_TQ_W, P_CF_TQ_B, P_CF_TK_W, P_CF_TK_B, P_CF_TV_W, P_CF_TV_B,
  P_CF_TMHA_IN_W, P_CF_TMHA_IN_B, P_CF_TMHA_OUT_W, P_CF_TMHA_OUT_B, P_CF_TO_W, P_CF_TO_B,
  P_CF_FLN_G, P_CF_FLN_B, P_CF_F_W, P_CF_F_B,
  P_R1_W, P_R1_B, P_R_LN_G, P_R_LN_B, P_R2_W, P_R2_B, P_R3_W, P_R3_B,
  P_AG1_W, P_AG1_B, P_AG2_W, P_AG2_B,
  P_SEQ_LN_G, P_SEQ_LN_B, P_VIS_LN_G, P_VIS_LN_B, P_SEM_LN_G, P_SEM_LN_B,
  P_GATING_W
};

extern "C" void kernel_launch(void* const* d_in, const int* in_sizes, int n_in,
                              void* d_out, int out_size, void* d_ws, size_t ws_size,
                              hipStream_t stream) {
  (void)in_sizes; (void)n_in; (void)out_size; (void)ws_size;

  auto F = [&](int i) { return (const float*)d_in[i]; };

  // workspace bump allocator
  char* ws = (char*)d_ws;
  size_t off = 0;
  auto alloc = [&](size_t bytes) -> void* {
    void* p = ws + off;
    off = (off + bytes + 255) & ~(size_t)255;
    return p;
  };

  // big ping-pong pool: 3 x [BL, D] f32
  float* bufA = (float*)alloc((size_t)BL_ * D_ * sizeof(float));
  float* bufB = (float*)alloc((size_t)BL_ * D_ * sizeof(float));
  float* bufC = (float*)alloc((size_t)BL_ * D_ * sizeof(float));

  // small f32 buffers
  float* tgt_vis = (float*)alloc((size_t)B_ * D_ * 4);
  float* qh      = (float*)alloc((size_t)B_ * AH_ * 4);
  float* agg     = (float*)alloc((size_t)B_ * D_ * 4);
  float* s1      = (float*)alloc((size_t)B_ * D_ * 4);
  float* s2      = (float*)alloc((size_t)B_ * D_ * 4);
  float* s3      = (float*)alloc((size_t)B_ * D_ * 4);
  float* vis_out = (float*)alloc((size_t)B_ * D_ * 4);
  float* sem_out = (float*)alloc((size_t)B_ * D_ * 4);
  float* seq_n   = (float*)alloc((size_t)B_ * D_ * 4);
  float* vis_n   = (float*)alloc((size_t)B_ * D_ * 4);
  float* sem_n   = (float*)alloc((size_t)B_ * D_ * 4);
  float* vo      = (float*)alloc((size_t)B_ * D_ * 4);
  float* to      = (float*)alloc((size_t)B_ * D_ * 4);
  float* cc      = (float*)alloc((size_t)B_ * 768 * 4);
  float* cc_ln   = (float*)alloc((size_t)B_ * 768 * 4);
  float* fusedb  = (float*)alloc((size_t)B_ * D_ * 4);
  float* ri      = (float*)alloc((size_t)B_ * 512 * 4);
  float* rh      = (float*)alloc((size_t)B_ * 128 * 4);
  float* rh2     = (float*)alloc((size_t)B_ * 128 * 4);
  float* h2      = (float*)alloc((size_t)B_ * 64 * 4);
  float* rwb     = (float*)alloc((size_t)B_ * 2 * 4);
  float* a1      = (float*)alloc((size_t)B_ * 64 * 4);
  float* agb     = (float*)alloc((size_t)B_ * 2 * 4);

  // convert weights to bf16 (per call; deterministic)
  auto cvt = [&](int idx, size_t n) -> const __bf16* {
    __bf16* dst = (__bf16*)alloc(n * sizeof(__bf16));
    f32_to_bf16_kernel<<<dim3((unsigned)((n + 255) / 256)), dim3(256), 0, stream>>>(F(idx), dst, (int)n);
    return dst;
  };
  const __bf16* w_vp      = cvt(P_VP_W,         (size_t)D_ * VD_);
  const __bf16* w_tp      = cvt(P_TP_W,         (size_t)D_ * TD_);
  const __bf16* w_ve_aq   = cvt(P_VE_AQ_W,      (size_t)AH_ * D_);
  const __bf16* w_ve_ak   = cvt(P_VE_AK_W,      (size_t)AH_ * D_);
  const __bf16* w_ve_out  = cvt(P_VE_OUT_W,     (size_t)D_ * D_);
  const __bf16* w_se_q    = cvt(P_SE_Q_W,       (size_t)D_ * D_);
  const __bf16* w_se_k    = cvt(P_SE_K_W,       (size_t)D_ * D_);
  const __bf16* w_se_v    = cvt(P_SE_V_W,       (size_t)D_ * D_);
  const __bf16* w_mha_in  = cvt(P_SE_MHA_IN_W,  (size_t)3 * D_ * D_);
  const __bf16* w_mha_out = cvt(P_SE_MHA_OUT_W, (size_t)D_ * D_);
  const __bf16* w_se_out  = cvt(P_SE_OUT_W,     (size_t)D_ * D_);
  const __bf16* w_cf_vv   = cvt(P_CF_VV_W,      (size_t)D_ * D_);
  const __bf16* w_cf_vin  = cvt(P_CF_VMHA_IN_W, (size_t)3 * D_ * D_);
  const __bf16* w_cf_vout = cvt(P_CF_VMHA_OUT_W,(size_t)D_ * D_);
  const __bf16* w_cf_vo   = cvt(P_CF_VO_W,      (size_t)D_ * D_);
  const __bf16* w_cf_tv   = cvt(P_CF_TV_W,      (size_t)D_ * D_);
  const __bf16* w_cf_tin  = cvt(P_CF_TMHA_IN_W, (size_t)3 * D_ * D_);
  const __bf16* w_cf_tout = cvt(P_CF_TMHA_OUT_W,(size_t)D_ * D_);
  const __bf16* w_cf_to   = cvt(P_CF_TO_W,      (size_t)D_ * D_);
  const __bf16* w_cf_f    = cvt(P_CF_F_W,       (size_t)D_ * 768);
  const __bf16* w_r1      = cvt(P_R1_W,         (size_t)128 * 512);
  const __bf16* w_r2      = cvt(P_R2_W,         (size_t)64 * 128);
  const __bf16* w_ag1     = cvt(P_AG1_W,        (size_t)64 * 768);

  auto gemm = [&](const float* A, const __bf16* W, const float* bias, float* C,
                  int M, int N, int K, int relu) {
    dim3 grid((unsigned)(M / 128), (unsigned)((N + 127) / 128));
    gemm_bf16_kernel<<<grid, 256, 0, stream>>>(A, W, bias, C, M, N, K, relu);
  };
  auto ln = [&](const float* x, int gi, int bi, float* y, int M, int Dd, int relu) {
    ln_kernel<<<dim3((unsigned)((M + 7) / 8)), 256, 0, stream>>>(x, F(gi), F(bi), y, M, Dd, relu);
  };

  // ---- multimodal projections -------------------------------------------
  gemm(F(I_MEM_VIS), w_vp, F(P_VP_B), bufA, BL_, D_, VD_, 0);     // mem_vis
  gemm(F(I_TGT_VIS), w_vp, F(P_VP_B), tgt_vis, B_, D_, VD_, 0);   // tgt_vis

  // ---- VisualExpert ------------------------------------------------------
  gemm(bufA, w_ve_ak, F(P_VE_AK_B), bufB, BL_, AH_, D_, 0);       // kh
  gemm(tgt_vis, w_ve_aq, F(P_VE_AQ_B), qh, B_, AH_, D_, 0);       // qh
  ve_attn_kernel<<<B_, 128, 0, stream>>>(qh, bufB, bufA, agg);    // agg
  gemm(agg, w_ve_out, F(P_VE_OUT_B), s1, B_, D_, D_, 0);
  ln(s1, P_VE_LN_G, P_VE_LN_B, vis_out, B_, D_, 1);               // relu(ln(..))

  // ---- SemanticExpert ----------------------------------------------------
  gemm(F(I_MEM_TXT), w_tp, F(P_TP_B), bufA, BL_, D_, TD_, 0);     // mem_txt (reuse A)
  gemm(bufA, w_se_k, F(P_SE_K_B), bufC, BL_, D_, D_, 0);          // sk
  gemm(bufC, w_mha_in + (size_t)D_ * D_, F(P_SE_MHA_IN_B) + D_,
       bufB, BL_, D_, D_, 0);                                     // kp
  gemm(bufA, w_se_v, F(P_SE_V_B), bufC, BL_, D_, D_, 0);          // sv (over sk)
  gemm(bufC, w_mha_in + (size_t)2 * D_ * D_, F(P_SE_MHA_IN_B) + 2 * D_,
       bufA, BL_, D_, D_, 0);                                     // vp (over mem_txt)
  gemm(F(I_TGT_EMB), w_se_q, F(P_SE_Q_B), s1, B_, D_, D_, 0);     // sq
  gemm(s1, w_mha_in, F(P_SE_MHA_IN_B), s2, B_, D_, D_, 0);        // qp
  se_attn_kernel<<<B_, 128, 0, stream>>>(s2, bufB, bufA, s3);     // attention core
  gemm(s3, w_mha_out, F(P_SE_MHA_OUT_B), s1, B_, D_, D_, 0);      // att
  ln(s1, P_SE_LN_G, P_SE_LN_B, s2, B_, D_, 0);
  gemm(s2, w_se_out, F(P_SE_OUT_B), sem_out, B_, D_, D_, 1);      // relu in epilogue

  // ---- branch layernorms -------------------------------------------------
  ln(F(I_SEQ), P_SEQ_LN_G, P_SEQ_LN_B, seq_n, B_, D_, 0);
  ln(vis_out, P_VIS_LN_G, P_VIS_LN_B, vis_n, B_, D_, 0);
  ln(sem_out, P_SEM_LN_G, P_SEM_LN_B, sem_n, B_, D_, 0);

  // ---- CrossModalFusion (Lk=1 -> attention == V-projection chain) -------
  gemm(vis_n, w_cf_vv, F(P_CF_VV_B), s1, B_, D_, D_, 0);
  gemm(s1, w_cf_vin + (size_t)2 * D_ * D_, F(P_CF_VMHA_IN_B) + 2 * D_, s2, B_, D_, D_, 0);
  gemm(s2, w_cf_vout, F(P_CF_VMHA_OUT_B), s3, B_, D_, D_, 0);
  gemm(s3, w_cf_vo, F(P_CF_VO_B), vo, B_, D_, D_, 0);
  gemm(sem_n, w_cf_tv, F(P_CF_TV_B), s1, B_, D_, D_, 0);
  gemm(s1, w_cf_tin + (size_t)2 * D_ * D_, F(P_CF_TMHA_IN_B) + 2 * D_, s2, B_, D_, D_, 0);
  gemm(s2, w_cf_tout, F(P_CF_TMHA_OUT_B), s3, B_, D_, D_, 0);
  gemm(s3, w_cf_to, F(P_CF_TO_B), to, B_, D_, D_, 0);

  concat3_kernel<<<dim3((B_ * 768 + 255) / 256), 256, 0, stream>>>(seq_n, vo, to, cc, B_);
  ln(cc, P_CF_FLN_G, P_CF_FLN_B, cc_ln, B_, 768, 0);
  gemm(cc_ln, w_cf_f, F(P_CF_F_B), fusedb, B_, D_, 768, 1);

  // ---- ItemCentricRouter -------------------------------------------------
  concat2_kernel<<<dim3((B_ * 512 + 255) / 256), 256, 0, stream>>>(F(I_TGT_EMB), tgt_vis, ri, B_);
  gemm(ri, w_r1, F(P_R1_B), rh, B_, 128, 512, 0);
  ln(rh, P_R_LN_G, P_R_LN_B, rh2, B_, 128, 1);
  gemm(rh2, w_r2, F(P_R2_B), h2, B_, 64, 128, 1);
  dual_softmax_kernel<<<dim3((B_ + 7) / 8), 256, 0, stream>>>(h2, F(P_R3_W), F(P_R3_B), rwb, B_, 64);

  // ---- align gating ------------------------------------------------------
  gemm(cc, w_ag1, F(P_AG1_B), a1, B_, 64, 768, 1);
  dual_softmax_kernel<<<dim3((B_ + 7) / 8), 256, 0, stream>>>(a1, F(P_AG2_W), F(P_AG2_B), agb, B_, 64);

  // ---- final combine -----------------------------------------------------
  combine_kernel<<<dim3((B_ * D_ + 255) / 256), 256, 0, stream>>>(
      F(I_SEQ), fusedb, vis_n, sem_n, rwb, agb, F(P_GATING_W), (float*)d_out, B_ * D_);
}
